// BiLevelRoutingAttention_89945205112996
// MI455X (gfx1250) — matrix operand order repaired and checked by reading, not compiled, not run on hardware
//
#include <hip/hip_runtime.h>

// ---------------------------------------------------------------------------
// BiLevelRoutingAttention for MI455X (gfx1250, wave32, WMMA bf16 -> f32 acc)
//  - q,k stored NHWC bf16  -> attention Q/K fragments load DIRECTLY from
//    global memory as two 16B global_load_b128 each (ISA fragment runs are
//    contiguous along the channel axis).
//  - v stored NCHW bf16    -> V B-fragments are two 16B row runs in global.
//  - GEMMs: double-buffered LDS, v_wmma_f32_16x16x32_bf16.
// ---------------------------------------------------------------------------

typedef __attribute__((ext_vector_type(16))) __bf16 v16bf;
typedef __attribute__((ext_vector_type(8)))  float  v8f;

#define NB     16      // batch
#define CCH    256     // channels
#define HPIX   56
#define WPIX   56
#define HWPIX  3136    // 56*56
#define NHEAD  8
#define DHEAD  32      // CCH / NHEAD
#define RS     8       // region side
#define RGRID  7       // 56/8
#define NREG   49      // 7*7
#define RS2    64      // tokens per region
#define TOPKV  4
#define NTILEN 49      // 3136 / 64 column tiles

__device__ __forceinline__ unsigned pack_bf16(float a, float b) {
    union { __bf16 h[2]; unsigned u; } x;
    x.h[0] = (__bf16)a; x.h[1] = (__bf16)b;
    return x.u;
}

// load a v16bf fragment as two 16-byte reads (works for LDS or global ptrs)
__device__ __forceinline__ v16bf load_frag(const __bf16* p0, const __bf16* p1) {
    union { v16bf v; uint4 q[2]; } u;
    u.q[0] = *reinterpret_cast<const uint4*>(p0);
    u.q[1] = *reinterpret_cast<const uint4*>(p1);
    return u.v;
}

// ---------------------------------------------------------------------------
// GEMM: C[OC x 3136] = W[OC x 256] * X[256 x 3136] + bias   (per batch)
// grid.x = (OC/64)*49, grid.y = batch, block = 128 (4 waves)
// MODE 0: QKV projection -> q,k as NHWC bf16; v as NCHW bf16
// MODE 1: output projection -> fout as NCHW f32
// ---------------------------------------------------------------------------
template <int MODE>
__global__ __launch_bounds__(128)
void gemm256_bias_kernel(const float* __restrict__ W,
                         const float* __restrict__ X,
                         const float* __restrict__ bias,
                         __bf16* __restrict__ qt,
                         __bf16* __restrict__ kt,
                         __bf16* __restrict__ vv,
                         float*  __restrict__ fout)
{
    const int n    = blockIdx.y;
    const int mt   = blockIdx.x / NTILEN;
    const int nt   = blockIdx.x % NTILEN;
    const int tid  = threadIdx.x;
    const int wave = tid >> 5;
    const int lane = tid & 31;
    const int h    = lane >> 4;
    const int rr   = lane & 15;

    // 80 B row stride: 16B aligned, bank-rotating; double buffered
    __shared__ __attribute__((aligned(16))) __bf16 As [2][64][40]; // W:  [M][K]
    __shared__ __attribute__((aligned(16))) __bf16 BsT[2][64][40]; // X^T:[N][K]

    const float* Xn = X + (size_t)n * CCH * HWPIX;
    const int m0   = mt * 64;
    const int col0 = nt * 64;

    auto load_stage = [&](int st, int k0) {
        for (int t = tid; t < 64 * 8; t += 128) {
            const int r = t >> 3, c4 = (t & 7) * 4;
            const float4 f = *reinterpret_cast<const float4*>(
                W + (size_t)(m0 + r) * CCH + k0 + c4);
            uint2 u; u.x = pack_bf16(f.x, f.y); u.y = pack_bf16(f.z, f.w);
            *reinterpret_cast<uint2*>(&As[st][r][c4]) = u;
        }
        for (int t = tid; t < 16 * 16; t += 128) {
            const int kp = t >> 4;
            const int c4 = (t & 15) * 4;
            const float* b0 = Xn + (size_t)(k0 + 2 * kp) * HWPIX + col0 + c4;
            const float4 r0 = *reinterpret_cast<const float4*>(b0);
            const float4 r1 = *reinterpret_cast<const float4*>(b0 + HWPIX);
            *reinterpret_cast<unsigned*>(&BsT[st][c4 + 0][2 * kp]) = pack_bf16(r0.x, r1.x);
            *reinterpret_cast<unsigned*>(&BsT[st][c4 + 1][2 * kp]) = pack_bf16(r0.y, r1.y);
            *reinterpret_cast<unsigned*>(&BsT[st][c4 + 2][2 * kp]) = pack_bf16(r0.z, r1.z);
            *reinterpret_cast<unsigned*>(&BsT[st][c4 + 3][2 * kp]) = pack_bf16(r0.w, r1.w);
        }
    };

    v8f acc[4];
    for (int i = 0; i < 4; ++i)
        for (int e = 0; e < 8; ++e) acc[i][e] = 0.0f;

    load_stage(0, 0);
    for (int ki = 0; ki < 8; ++ki) {
        __syncthreads();
        if (ki < 7) load_stage((ki + 1) & 1, (ki + 1) * 32);
        const int st = ki & 1;
        const __bf16* arow = &As[st][wave * 16 + rr][0];
        const v16bf afrag = load_frag(arow + 8 * h, arow + 16 + 8 * h);
        for (int nc = 0; nc < 4; ++nc) {
            const __bf16* brow = &BsT[st][nc * 16 + rr][0];
            const v16bf bfrag = load_frag(brow + 8 * h, brow + 16 + 8 * h);
            acc[nc] = __builtin_amdgcn_wmma_f32_16x16x32_bf16(
                false, afrag, false, bfrag, (short)0, acc[nc], false, false);
        }
    }

    // Epilogue: lane holds rows oc0..oc0+7 (contiguous channels), col = pix
    const int oc0 = m0 + wave * 16 + 8 * h;
    for (int nc = 0; nc < 4; ++nc) {
        const int pix = col0 + nc * 16 + rr;
        if (MODE == 0) {
            const int sel = oc0 >> 8;            // 0=q, 1=k, 2=v (tile-uniform)
            const int ch0 = oc0 & 255;
            if (sel < 2) {                       // NHWC bf16, one 16B store
                union { __bf16 hh[8]; uint4 u; } pk;
                for (int r = 0; r < 8; ++r)
                    pk.hh[r] = (__bf16)(acc[nc][r] + bias[oc0 + r]);
                __bf16* dst = (sel == 0 ? qt : kt)
                              + ((size_t)n * HWPIX + pix) * CCH + ch0;
                *reinterpret_cast<uint4*>(dst) = pk.u;
            } else {                             // v: NCHW bf16
                for (int r = 0; r < 8; ++r)
                    vv[((size_t)n * CCH + ch0 + r) * HWPIX + pix] =
                        (__bf16)(acc[nc][r] + bias[oc0 + r]);
            }
        } else {                                 // NCHW f32
            for (int r = 0; r < 8; ++r)
                fout[((size_t)n * CCH + oc0 + r) * HWPIX + pix] =
                    acc[nc][r] + bias[oc0 + r];
        }
    }
}

// ---------------------------------------------------------------------------
// Kernel 2: per-region 8x8 means of q,k (NHWC bf16) -> q_r,k_r (n, reg, c) f32
// consecutive threads = consecutive channels -> coalesced u16 loads
// ---------------------------------------------------------------------------
__global__ void region_mean_kernel(const __bf16* __restrict__ qt,
                                   const __bf16* __restrict__ kt,
                                   float* __restrict__ qr,
                                   float* __restrict__ kr)
{
    const int idx = blockIdx.x * blockDim.x + threadIdx.x;
    if (idx >= NB * NREG * CCH) return;
    const int c   = idx & 255;
    const int reg = (idx >> 8) % NREG;
    const int n   = idx / (NREG * CCH);
    const int rH  = reg / RGRID, rW = reg % RGRID;
    float sq = 0.f, sk = 0.f;
    for (int p = 0; p < RS; ++p)
        for (int w = 0; w < RS; ++w) {
            const size_t off = ((size_t)n * HWPIX
                                + (rH * RS + p) * WPIX + rW * RS + w) * CCH + c;
            sq += (float)qt[off];
            sk += (float)kt[off];
        }
    qr[idx] = sq * (1.0f / 64.0f);
    kr[idx] = sk * (1.0f / 64.0f);
}

// ---------------------------------------------------------------------------
// Kernel 3: routing scores a_r[n,i,:] (49 dots of length 256) + serial top-4
// (strict '>' => ties pick the lowest index, matching jax.lax.top_k)
// ---------------------------------------------------------------------------
__global__ __launch_bounds__(64)
void route_topk_kernel(const float* __restrict__ qr,
                       const float* __restrict__ kr,
                       int* __restrict__ idxr)
{
    const int n = blockIdx.x / NREG;
    const int i = blockIdx.x % NREG;
    const int j = threadIdx.x;
    __shared__ float s[NREG];
    if (j < NREG) {
        const float* a = qr + ((size_t)n * NREG + i) * CCH;
        const float* b = kr + ((size_t)n * NREG + j) * CCH;
        float dot = 0.f;
        for (int c = 0; c < CCH; ++c) dot += a[c] * b[c];
        s[j] = dot;
    }
    __syncthreads();
    if (j == 0) {
        bool used[NREG];
        for (int t = 0; t < NREG; ++t) used[t] = false;
        for (int t = 0; t < TOPKV; ++t) {
            float best = -3.4e38f; int bi = 0;
            for (int jj = 0; jj < NREG; ++jj)
                if (!used[jj] && s[jj] > best) { best = s[jj]; bi = jj; }
            used[bi] = true;
            idxr[((size_t)n * NREG + i) * TOPKV + t] = bi;
        }
    }
}

// ---------------------------------------------------------------------------
// Kernel 4: flash-style routed attention per (n, head, region).
// Q/K fragments: two global_load_b128 straight from NHWC bf16 (no LDS stage).
// V fragments:   two global_load_b128 from NCHW bf16 rows.
// LDS only holds the 64x64 score tile + P + row stats (~27 KB).
// ---------------------------------------------------------------------------
__global__ __launch_bounds__(128)
void routed_attention_kernel(const __bf16* __restrict__ qt,
                             const __bf16* __restrict__ kt,
                             const __bf16* __restrict__ vv,
                             const int* __restrict__ idxr,
                             float* __restrict__ out)
{
    const int blk  = blockIdx.x;
    const int reg  = blk % NREG;
    const int head = (blk / NREG) % NHEAD;
    const int n    = blk / (NREG * NHEAD);
    const int tid  = threadIdx.x;
    const int wave = tid >> 5;
    const int lane = tid & 31;
    const int h    = lane >> 4;
    const int rr   = lane & 15;

    const int rH = reg / RGRID, rW = reg % RGRID;
    const float scale = 0.17677669529663687f;      // 32^-0.5

    __shared__ __attribute__((aligned(16))) float  Sb[RS2][68];
    __shared__ __attribute__((aligned(16))) __bf16 Pb[RS2][72];
    __shared__ float rowM[RS2], rowL[RS2], rowA[RS2];

    const __bf16* qbase = qt + (size_t)n * HWPIX * CCH + head * DHEAD;
    const __bf16* kbase = kt + (size_t)n * HWPIX * CCH + head * DHEAD;

    // ---- Q A-fragment straight from global (held in registers all along)
    const int tq   = wave * 16 + rr;
    const int pixq = (rH * RS + (tq >> 3)) * WPIX + rW * RS + (tq & 7);
    const __bf16* qp = qbase + (size_t)pixq * CCH;
    const v16bf afrag = load_frag(qp + 8 * h, qp + 16 + 8 * h);

    if (tid < RS2) { rowM[tid] = -3.4e38f; rowL[tid] = 0.f; }

    v8f oacc[2];
    for (int i = 0; i < 2; ++i)
        for (int e = 0; e < 8; ++e) oacc[i][e] = 0.0f;

    for (int kk = 0; kk < TOPKV; ++kk) {
        const int regj = idxr[((size_t)n * NREG + reg) * TOPKV + kk];
        const int rHj = regj / RGRID, rWj = regj % RGRID;

        __syncthreads();       // previous iteration's Pb fully consumed

        // ---- S = Q * K^T  (64x64, scale folded into the store)
        for (int nt = 0; nt < 4; ++nt) {
            const int tk   = nt * 16 + rr;
            const int pixk = (rHj * RS + (tk >> 3)) * WPIX + rWj * RS + (tk & 7);
            const __bf16* kp = kbase + (size_t)pixk * CCH;
            const v16bf bfrag = load_frag(kp + 8 * h, kp + 16 + 8 * h);
            v8f sc;
            for (int e = 0; e < 8; ++e) sc[e] = 0.0f;
            sc = __builtin_amdgcn_wmma_f32_16x16x32_bf16(
                false, afrag, false, bfrag, (short)0, sc, false, false);
            for (int r = 0; r < 8; ++r)
                Sb[wave * 16 + r + 8 * h][nt * 16 + rr] = sc[r] * scale;
        }
        __syncthreads();

        // ---- online softmax row update (one thread per row)
        if (tid < RS2) {
            const float mOld = rowM[tid];
            float mNew = mOld;
            for (int j = 0; j < RS2; ++j) mNew = fmaxf(mNew, Sb[tid][j]);
            const float alpha = expf(mOld - mNew);
            float sum = 0.f;
            for (int j = 0; j < RS2; ++j) {
                const float p = expf(Sb[tid][j] - mNew);
                Pb[tid][j] = (__bf16)p;
                sum += p;
            }
            rowM[tid] = mNew;
            rowL[tid] = rowL[tid] * alpha + sum;
            rowA[tid] = alpha;
        }
        __syncthreads();

        // ---- rescale O, then O += P * V   (K = 64 -> two K steps)
        for (int i = 0; i < 2; ++i)
            for (int r = 0; r < 8; ++r)
                oacc[i][r] *= rowA[wave * 16 + r + 8 * h];
        const __bf16* prow = &Pb[wave * 16 + rr][0];
        for (int ks = 0; ks < 2; ++ks) {
            const v16bf pfrag = load_frag(prow + ks * 32 + 16 * h,
                                          prow + ks * 32 + 16 * h + 8);
            for (int nt = 0; nt < 2; ++nt) {
                const int dd = nt * 16 + rr;
                const __bf16* vch = vv + ((size_t)n * CCH + head * DHEAD + dd) * HWPIX;
                const int prow0 = (ks * 32 + 16 * h) >> 3;     // region row
                const __bf16* vp0 = vch + (rHj * RS + prow0) * WPIX + rWj * RS;
                const v16bf vfrag = load_frag(vp0, vp0 + WPIX);
                oacc[nt] = __builtin_amdgcn_wmma_f32_16x16x32_bf16(
                    false, pfrag, false, vfrag, (short)0, oacc[nt], false, false);
            }
        }
    }
    __syncthreads();

    // ---- epilogue: O /= l, scatter back to (n, c, h, w) f32
    for (int nt = 0; nt < 2; ++nt) {
        const int dd = nt * 16 + rr;
        for (int r = 0; r < 8; ++r) {
            const int row = wave * 16 + r + 8 * h;
            const float val = oacc[nt][r] / rowL[row];
            const int hh = rH * RS + (row >> 3);
            const int ww = rW * RS + (row & 7);
            out[((size_t)n * CCH + head * DHEAD + dd) * HWPIX + hh * WPIX + ww] = val;
        }
    }
}

// ---------------------------------------------------------------------------
// Kernel 5: depthwise 3x3 LEPE on v (NCHW bf16) + b_lepe, added into att (f32)
// ---------------------------------------------------------------------------
__global__ void lepe_add_kernel(const __bf16* __restrict__ v,
                                const float* __restrict__ wl,
                                const float* __restrict__ bl,
                                float* __restrict__ out)
{
    const size_t idx = (size_t)blockIdx.x * blockDim.x + threadIdx.x;
    if (idx >= (size_t)NB * CCH * HWPIX) return;
    const int pix = (int)(idx % HWPIX);
    const int c   = (int)((idx / HWPIX) % CCH);
    const int hh  = pix / WPIX, ww = pix % WPIX;
    const __bf16* vp = v + (idx - pix);                 // channel plane base
    const float* wc = wl + c * 9;
    float s = 0.f;
    for (int dy = -1; dy <= 1; ++dy) {
        const int y = hh + dy;
        if (y < 0 || y >= HPIX) continue;
        for (int dx = -1; dx <= 1; ++dx) {
            const int x = ww + dx;
            if (x < 0 || x >= WPIX) continue;
            s += (float)vp[y * WPIX + x] * wc[(dy + 1) * 3 + (dx + 1)];
        }
    }
    out[idx] = out[idx] + s + bl[c];
}

// ---------------------------------------------------------------------------
extern "C" void kernel_launch(void* const* d_in, const int* in_sizes, int n_in,
                              void* d_out, int out_size, void* d_ws, size_t ws_size,
                              hipStream_t stream)
{
    const float* x      = (const float*)d_in[0];
    const float* w_qkv  = (const float*)d_in[1];
    const float* b_qkv  = (const float*)d_in[2];
    const float* w_lepe = (const float*)d_in[3];
    const float* b_lepe = (const float*)d_in[4];
    const float* w_out  = (const float*)d_in[5];
    const float* b_out  = (const float*)d_in[6];
    float* out = (float*)d_out;

    const size_t QE = (size_t)NB * CCH * HWPIX;         // 12,845,056 elements
    char* base = (char*)d_ws;
    __bf16* qt  = (__bf16*)(base);                      // NHWC bf16
    __bf16* kt  = (__bf16*)(base + QE * 2);             // NHWC bf16
    __bf16* vv  = (__bf16*)(base + QE * 4);             // NCHW bf16
    float*  att = (float*) (base + QE * 6);             // NCHW f32
    float*  qr  = (float*) (base + QE * 10);
    float*  kr  = qr + (size_t)NB * NREG * CCH;
    int*    idxr = (int*)(kr + (size_t)NB * NREG * CCH);

    // 1) QKV projection (M = 768) -> bf16 q/k (NHWC), v (NCHW)
    dim3 g1(12 * NTILEN, NB);
    gemm256_bias_kernel<0><<<g1, 128, 0, stream>>>(w_qkv, x, b_qkv,
                                                   qt, kt, vv, nullptr);

    // 2) region means
    region_mean_kernel<<<(NB * NREG * CCH + 255) / 256, 256, 0, stream>>>(qt, kt, qr, kr);

    // 3) routing + top-4
    route_topk_kernel<<<NB * NREG, 64, 0, stream>>>(qr, kr, idxr);

    // 4) routed flash attention
    routed_attention_kernel<<<NB * NHEAD * NREG, 128, 0, stream>>>(qt, kt, vv, idxr, att);

    // 5) LEPE + bias added into attention output
    lepe_add_kernel<<<(int)((QE + 255) / 256), 256, 0, stream>>>(vv, w_lepe, b_lepe, att);

    // 6) output projection (M = 256) -> f32 out
    dim3 g6(4 * NTILEN, NB);
    gemm256_bias_kernel<1><<<g6, 128, 0, stream>>>(w_out, att, b_out,
                                                   nullptr, nullptr, nullptr, out);
}
